// ClusterAwareBatchNorm2d_14250701488155
// MI455X (gfx1250) — matrix operand order repaired and checked by reading, not compiled, not run on hardware
//
#include <hip/hip_runtime.h>
#include <hip/hip_bf16.h>

// ---------------------------------------------------------------------------
// ClusterAwareBatchNorm2d for MI455X (gfx1250, wave32).
// B=64, C=256, H=W=56 (3136 px per plane, = 784 float4), fp32 in/out.
//
// Pipeline (all on `stream`, no host sync):
//   1) stats_kernel        : per-(b,c) mean + unbiased var     (1 pass over x)
//   2) rownorm_kernel      : row-normalize mu -> fp16
//   3) gram_kernel         : G = nrm @ nrm^T  via V_WMMA_F32_16X16X32_F16
//   4) finch_kernel        : 1-NN + connected components -> labels, denom
//   5) cluster_stats_kernel: per-cluster mu / sigma2 aggregation
//   6) scale_shift_kernel  : per-(b,c) affine coefficients
//   7) apply_kernel        : out = x*scale + shift             (1 pass over x)
// ---------------------------------------------------------------------------

#define BN 64
#define CN 256
#define HW 3136      // 56*56
#define HW4 784      // HW/4
#define EPSV 1e-5f

typedef __attribute__((ext_vector_type(16))) _Float16 v16h;
typedef __attribute__((ext_vector_type(8)))  float    v8f;
typedef __attribute__((ext_vector_type(4)))  float    fvec4;   // native vector: OK for nontemporal builtins

// --- 1) per-plane mean / unbiased variance ---------------------------------
__global__ void stats_kernel(const float* __restrict__ x,
                             float* __restrict__ mu,
                             float* __restrict__ sig2) {
    const int plane = blockIdx.x;                 // b*C + c
    const fvec4* xp = (const fvec4*)x + (size_t)plane * HW4;
    float s = 0.f, ss = 0.f;
    for (int i = threadIdx.x; i < HW4; i += blockDim.x) {
        fvec4 v = xp[i];
        s  += v.x + v.y + v.z + v.w;
        ss += v.x*v.x + v.y*v.y + v.z*v.z + v.w*v.w;
    }
    __shared__ float sh[256], sh2[256];
    sh[threadIdx.x] = s; sh2[threadIdx.x] = ss;
    __syncthreads();
    for (int off = 128; off > 0; off >>= 1) {
        if ((int)threadIdx.x < off) {
            sh[threadIdx.x]  += sh[threadIdx.x + off];
            sh2[threadIdx.x] += sh2[threadIdx.x + off];
        }
        __syncthreads();
    }
    if (threadIdx.x == 0) {
        float m = sh[0] * (1.0f / (float)HW);
        mu[plane]   = m;
        sig2[plane] = (sh2[0] - sh[0] * m) * (1.0f / (float)(HW - 1));  // ddof=1
    }
}

// --- 2) normalize rows of mu, convert to fp16 ------------------------------
__global__ void rownorm_kernel(const float* __restrict__ mu,
                               _Float16* __restrict__ nrm) {
    const int b = blockIdx.x;
    const int c = threadIdx.x;                    // 256 threads
    __shared__ float sh[256];
    __shared__ float rn;
    float v = mu[b * CN + c];
    sh[c] = v * v;
    __syncthreads();
    for (int off = 128; off > 0; off >>= 1) {
        if (c < off) sh[c] += sh[c + off];
        __syncthreads();
    }
    if (c == 0) rn = rsqrtf(sh[0]);
    __syncthreads();
    nrm[b * CN + c] = (_Float16)(v * rn);
}

// --- 3) gram matrix G[64,64] = nrm @ nrm^T via WMMA ------------------------
// One wave per 16x16 tile; 16 waves (512 threads) cover the 4x4 tile grid.
// K loop: 256 in chunks of 32 -> 8 v_wmma_f32_16x16x32_f16 per wave.
__global__ void gram_kernel(const _Float16* __restrict__ nrm,
                            float* __restrict__ G) {
    const int wave = threadIdx.x >> 5;            // 0..15
    const int lane = threadIdx.x & 31;
    const int h    = lane >> 4;                   // lane half selects K-halves
    const int lm   = lane & 15;                   // M (for A) / N (for B)
    const int m0   = (wave >> 2) * 16;
    const int n0   = (wave & 3) * 16;

    v8f c = {};
    for (int kc = 0; kc < CN; kc += 32) {
        const _Float16* ra = nrm + (m0 + lm) * CN + kc;
        const _Float16* rb = nrm + (n0 + lm) * CN + kc;
        v16h a, b;
        // A 16x32 fp16, contiguous k order used here: lane half 0 -> K = 0..15,
        //                      lane half 1 -> K = 16..31.
        #pragma unroll
        for (int e = 0; e < 8; ++e) {
            a[e]     = ra[h * 16 + e];
            a[8 + e] = ra[h * 16 + 8 + e];
        }
        // B 32x16 fp16, same contiguous k order as A: lanes 0-15 hold K=0..15, lanes 16-31 K=16..31.
        #pragma unroll
        for (int e = 0; e < 16; ++e) b[e] = rb[h * 16 + e];

        c = __builtin_amdgcn_wmma_f32_16x16x32_f16(
                /*neg_a=*/false, a, /*neg_b=*/false, b,
                /*c_mod=*/(short)0, c, /*reuse_a=*/false, /*reuse_b=*/false);
    }
    // C/D layout: VGPR r -> row M = r + 8*half, col N = lane%16.
    #pragma unroll
    for (int r = 0; r < 8; ++r)
        G[(m0 + h * 8 + r) * BN + n0 + lm] = c[r];
}

// --- 4) FINCH first partition: 1-NN + weak connected components ------------
__global__ void finch_kernel(const float* __restrict__ G,
                             int* __restrict__ labels_out,
                             float* __restrict__ denom_out) {
    __shared__ int nnS[BN];
    __shared__ int lab[BN];
    __shared__ int labNew[BN];
    const int i = threadIdx.x;                    // 64 threads

    // first nearest neighbor (argmin of cosine distance, first occurrence)
    float best = __builtin_inff();
    int bestj = 0;
    for (int j = 0; j < BN; ++j) {
        if (j == i) continue;
        float d = 1.0f - G[i * BN + j];
        if (d < best) { best = d; bestj = j; }
    }
    nnS[i] = bestj;
    lab[i] = i;
    __syncthreads();

    // adjacency: j==nn[i] || nn[j]==i || nn[i]==nn[j]  (diag zeroed)
    for (int t = 0; t < BN; ++t) {
        int m  = lab[i];
        int ni = nnS[i];
        for (int j = 0; j < BN; ++j) {
            if (j == i) continue;
            if (ni == j || nnS[j] == i || ni == nnS[j]) {
                int lj = lab[j];
                m = (lj < m) ? lj : m;
            }
        }
        labNew[i] = m;
        __syncthreads();
        lab[i] = labNew[i];
        __syncthreads();
    }
    labels_out[i] = lab[i];
    int cnt = 0;
    for (int b = 0; b < BN; ++b) cnt += (lab[b] == i) ? 1 : 0;
    denom_out[i] = (float)cnt + EPSV;
}

// --- 5) per-cluster statistics --------------------------------------------
__global__ void cluster_stats_kernel(const float* __restrict__ mu,
                                     const float* __restrict__ sig2,
                                     const int*   __restrict__ labels,
                                     const float* __restrict__ denom,
                                     float* __restrict__ cmu,
                                     float* __restrict__ cs2) {
    const int c = blockIdx.x;                     // 256 blocks
    const int l = threadIdx.x;                    // 64 threads (cluster id)
    __shared__ float mcol[BN], scol[BN];
    __shared__ int   slab[BN];
    mcol[l] = mu[l * CN + c];
    scol[l] = sig2[l * CN + c];
    slab[l] = labels[l];
    __syncthreads();

    const float d = denom[l];
    float sm = 0.f, sv = 0.f;
    for (int b = 0; b < BN; ++b)
        if (slab[b] == l) { sm += mcol[b]; sv += scol[b]; }
    const float cm = sm / d;
    float diff2 = 0.f;
    for (int b = 0; b < BN; ++b)
        if (slab[b] == l) { float t = mcol[b] - cm; diff2 += t * t; }
    cmu[l * CN + c] = cm;
    cs2[l * CN + c] = sv / d + diff2 / d;
}

// --- 6) per-(b,c) affine coefficients --------------------------------------
__global__ void scale_shift_kernel(const float* __restrict__ cmu,
                                   const float* __restrict__ cs2,
                                   const int*   __restrict__ labels,
                                   const float* __restrict__ rmean,
                                   const float* __restrict__ rvar,
                                   const float* __restrict__ weight,
                                   const float* __restrict__ bias,
                                   const float* __restrict__ sr_p,
                                   float* __restrict__ scale,
                                   float* __restrict__ shift) {
    const int i = blockIdx.x * blockDim.x + threadIdx.x;   // < B*C
    const int b = i >> 8;
    const int c = i & 255;
    const float sr = fminf(fmaxf(sr_p[0], 0.f), 1.f);
    const float rt = 1.f - sr;
    const int l = labels[b];
    const float varm = rt * cs2[l * CN + c] + sr * rvar[c] + EPSV;
    const float mum  = rt * cmu[l * CN + c] + sr * rmean[c];
    const float s = weight[c] * rsqrtf(varm);
    scale[i] = s;
    shift[i] = bias[c] - s * mum;
}

// --- 7) streaming apply pass (bandwidth bound) -----------------------------
__global__ void apply_kernel(const float* __restrict__ x,
                             const float* __restrict__ scale,
                             const float* __restrict__ shift,
                             float* __restrict__ out) {
    const int plane = blockIdx.x;                 // b*C + c
    const float s  = scale[plane];
    const float sh = shift[plane];
    const fvec4* xp = (const fvec4*)x + (size_t)plane * HW4;
    fvec4* op = (fvec4*)out + (size_t)plane * HW4;
    for (int i = threadIdx.x; i < HW4; i += blockDim.x) {
        fvec4 v = xp[i];
        v.x = v.x * s + sh;
        v.y = v.y * s + sh;
        v.z = v.z * s + sh;
        v.w = v.w * s + sh;
        __builtin_nontemporal_store(v, &op[i]);   // write-once stream, skip L2 residency
    }
}

// ---------------------------------------------------------------------------
extern "C" void kernel_launch(void* const* d_in, const int* in_sizes, int n_in,
                              void* d_out, int out_size, void* d_ws, size_t ws_size,
                              hipStream_t stream) {
    (void)in_sizes; (void)n_in; (void)out_size; (void)ws_size;
    const float* x      = (const float*)d_in[0];
    const float* rmean  = (const float*)d_in[1];
    const float* rvar   = (const float*)d_in[2];
    const float* weight = (const float*)d_in[3];
    const float* bias   = (const float*)d_in[4];
    const float* srp    = (const float*)d_in[5];

    // Workspace layout (float units). Total ~443 KB.
    float* ws      = (float*)d_ws;
    float* mu      = ws;                   // 16384
    float* sig2    = ws + 16384;           // 16384
    float* cmu     = ws + 32768;           // 16384
    float* cs2     = ws + 49152;           // 16384
    float* scale   = ws + 65536;           // 16384
    float* shift   = ws + 81920;           // 16384
    float* G       = ws + 98304;           // 4096
    float* denom   = ws + 102400;          // 64
    _Float16* nrm  = (_Float16*)(ws + 102464);  // 16384 halves (32B aligned)
    int* labels    = (int*)(ws + 110656);  // 64

    stats_kernel        <<<BN * CN, 256, 0, stream>>>(x, mu, sig2);
    rownorm_kernel      <<<BN, 256, 0, stream>>>(mu, nrm);
    gram_kernel         <<<1, 512, 0, stream>>>(nrm, G);     // 16 waves, WMMA
    finch_kernel        <<<1, 64, 0, stream>>>(G, labels, denom);
    cluster_stats_kernel<<<CN, 64, 0, stream>>>(mu, sig2, labels, denom, cmu, cs2);
    scale_shift_kernel  <<<(BN * CN) / 256, 256, 0, stream>>>(cmu, cs2, labels,
                                                              rmean, rvar, weight,
                                                              bias, srp, scale, shift);
    apply_kernel        <<<BN * CN, 256, 0, stream>>>(x, scale, shift, (float*)d_out);
}
